// SLiM_CZ_V1_80848464380173
// MI455X (gfx1250) — matrix operand order repaired and checked by reading, not compiled, not run on hardware
//
#include <hip/hip_runtime.h>
#include <math.h>

// ---------------------------------------------------------------------------
// GPT forward for MI455X (gfx1250): bf16 WMMA GEMM core + TDM async staging.
// B=2, T=1024, D=1024, H=16, HD=64, L=4, F=4096, V=32000
//
// GEMM: 64x64 block tile, 8 waves (4M x 2N), each wave a 16x32 strip
//       -> 2 v_wmma_f32_16x16x32_bf16 per wave per 32-deep k-step.
// TRANS_B GEMMs (scores, logits) stage the B tile with the Tensor Data Mover
// (tensor_load_to_lds, double-buffered, TDM pad for LDS banking); the rest
// use vectorized global->LDS staging with fragment-order layouts so each
// lane's 16 bf16 fragment elements are one contiguous 32B ds_load run.
// ---------------------------------------------------------------------------

typedef __attribute__((ext_vector_type(16))) __bf16 v16bf;
typedef __attribute__((ext_vector_type(8)))  __bf16 v8bf;
typedef __attribute__((ext_vector_type(4)))  __bf16 v4bf;
typedef __attribute__((ext_vector_type(8)))  float  v8f;
typedef __attribute__((ext_vector_type(4)))  float  v4f;
typedef __attribute__((ext_vector_type(4)))  unsigned int u32x4;
typedef __attribute__((ext_vector_type(4)))  int          i32x4;
typedef __attribute__((ext_vector_type(8)))  int          i32x8;

#define BLK 256
#define TM  64
#define TN  64
#define TK  32
#define TKP 48   // padded bf16 LDS row: 96B keeps 32B alignment
#define TFP 40   // padded f32 TDM row: 128B tile row + 32B pad = 160B

// ---- Tensor Data Mover: 2D f32 tile load, global -> LDS (D# per ISA 8.3/8.4)
__device__ __forceinline__ void tdm_load_tile_f32(
    const float* gsrc, void* ldsdst,
    int tile_k, int tile_n, int tens_k, int tens_n, int stride_elems)
{
    unsigned long long ga = (unsigned long long)(const void*)gsrc;
    unsigned lds_off = (unsigned)(unsigned long long)ldsdst;
    u32x4 g0;
    g0[0] = 1u;                                                 // count=1 (valid)
    g0[1] = lds_off;                                            // lds_addr
    g0[2] = (unsigned)ga;                                       // global_addr[31:0]
    g0[3] = ((unsigned)(ga >> 32) & 0x01FFFFFFu) | (2u << 30);  // addr[56:32] | type=2
    unsigned w0 = (2u << 16)      // data_size = 4B
                | (1u << 20)      // pad_enable
                | (4u << 22)      // pad_interval: 32 dwords (one 128B tile row)
                | (7u << 25);     // pad_amount: 8 dwords (32B) -> 160B LDS rows
    unsigned w1 = ((unsigned)tens_k & 0xFFFFu) << 16;                       // dim0[15:0]
    unsigned w2 = ((unsigned)tens_k >> 16) | (((unsigned)tens_n & 0xFFFFu) << 16);
    unsigned w3 = ((unsigned)tens_n >> 16) | ((unsigned)tile_k << 16);      // tile_dim0
    unsigned w4 = (unsigned)tile_n;                                         // tile_dim1
    unsigned w5 = (unsigned)stride_elems;                                   // dim0_stride
    i32x8 g1 = { (int)w0, (int)w1, (int)w2, (int)w3, (int)w4, (int)w5, 0, 0 };
#if __has_builtin(__builtin_amdgcn_tensor_load_to_lds)
    i32x4 z4 = {0, 0, 0, 0};
    i32x8 z8 = {0, 0, 0, 0, 0, 0, 0, 0};
    // amdgpu-toolchain (clang-23) 6-arg form:
    // (u32x4 g0, i32x8 g1, i32x4 g2, i32x4 g3, i32x8, i32 cpol)
    __builtin_amdgcn_tensor_load_to_lds(g0, g1, z4, z4, z8, 0);
#else
    asm volatile("tensor_load_to_lds %0, %1" :: "s"(g0), "s"(g1) : "memory");
#endif
}

template<int Ncnt>
__device__ __forceinline__ void wait_tensorcnt()
{
#if __has_builtin(__builtin_amdgcn_s_wait_tensorcnt)
    __builtin_amdgcn_s_wait_tensorcnt((short)Ncnt);
#else
    if constexpr (Ncnt == 0) asm volatile("s_wait_tensorcnt 0x0" ::: "memory");
    else                     asm volatile("s_wait_tensorcnt 0x1" ::: "memory");
#endif
}

// EPI: 0 = plain (+bias)           (HAS_RES: += residual)
//      1 = GELU(acc + bias)
//      2 = causal mask + scale (scores):  n<=m ? acc*scale : -1e9
template<int EPI, bool TRANS_B, bool HAS_RES>
__global__ __launch_bounds__(BLK) void gemm_wmma_k(
    const float* __restrict__ A, const float* __restrict__ W,
    const float* __restrict__ bias, const float* __restrict__ res,
    float* __restrict__ C,
    int M, int N, int K, int lda, int ldb, int ldc, int Hn,
    long sAb, long sAh, long sWb, long sWh, long sCb, long sCh,
    float scale)
{
    // A staged k-contiguous: position = k
    // so lane(half) reads As2[row][16*half .. +15] contiguously (32B).
    __shared__ alignas(32) __bf16 As2[TM][TKP];

    const int bb = blockIdx.z / Hn;
    const int hh = blockIdx.z % Hn;
    const float* Ab = A + (long)bb * sAb + (long)hh * sAh;
    const float* Wb = W + (long)bb * sWb + (long)hh * sWh;
    float*       Cb = C + (long)bb * sCb + (long)hh * sCh;
    const float* Rb = HAS_RES ? (res + (long)bb * sCb + (long)hh * sCh) : nullptr;

    const int mBase = blockIdx.y * TM;
    const int nBase = blockIdx.x * TN;
    const int tid   = threadIdx.x;
    const int lane  = tid & 31;
    const int wave  = tid >> 5;     // 0..7
    const int wm    = wave >> 1;    // 0..3  (16-row strip)
    const int wn    = wave & 1;     // 0..1  (32-col strip)
    const int half  = lane >> 4;
    const int ln    = lane & 15;

    v8f acc[2] = {};

    auto stageA = [&](int k0) {
        #pragma unroll
        for (int i = 0; i < 2; ++i) {
            int idx = tid + i * BLK;            // 0..511
            int r   = idx >> 3;                 // 0..63
            int cb  = (idx & 7) * 4;            // 0,4,..,28
            v4f a   = *(const v4f*)&Ab[(long)(mBase + r) * lda + k0 + cb];
            v4bf ab = __builtin_convertvector(a, v4bf);
            int p   = cb;
            *(v4bf*)&As2[r][p] = ab;
        }
        if (k0 + TK < K)
            __builtin_prefetch(&Ab[(long)(mBase + (tid >> 3)) * lda + k0 + TK + (tid & 7) * 4], 0, 3);
    };

    if constexpr (TRANS_B) {
        // ---- B tile via Tensor Data Mover, double buffered ----
        __shared__ alignas(32) float Bf[2][TN][TFP];
        if (wave == 0)
            tdm_load_tile_f32(&Wb[(long)nBase * ldb], &Bf[0][0][0], TK, TN, K, N, ldb);

        for (int k0 = 0; k0 < K; k0 += TK) {
            const int cur = (k0 / TK) & 1;
            stageA(k0);
            if (wave == 0) {
                if (k0 + TK < K) {
                    tdm_load_tile_f32(&Wb[(long)nBase * ldb + (k0 + TK)],
                                      &Bf[cur ^ 1][0][0], TK, TN, K, N, ldb);
                    wait_tensorcnt<1>();   // current chunk complete
                } else {
                    wait_tensorcnt<0>();
                }
            }
            __syncthreads();

            const v16bf af = *(const v16bf*)&As2[wm * 16 + ln][16 * half];
            const float* bp0 = &Bf[cur][wn * 32 + ln][16 * half];
            const float* bp1 = &Bf[cur][wn * 32 + 16 + ln][16 * half];
            v8bf b0lo = __builtin_convertvector(*(const v8f*)bp0, v8bf);
            v8bf b0hi = __builtin_convertvector(*(const v8f*)(bp0 + 8), v8bf);
            v8bf b1lo = __builtin_convertvector(*(const v8f*)bp1, v8bf);
            v8bf b1hi = __builtin_convertvector(*(const v8f*)(bp1 + 8), v8bf);
            v16bf b0 = __builtin_shufflevector(b0lo, b0hi,
                        0,1,2,3,4,5,6,7,8,9,10,11,12,13,14,15);
            v16bf b1 = __builtin_shufflevector(b1lo, b1hi,
                        0,1,2,3,4,5,6,7,8,9,10,11,12,13,14,15);

            acc[0] = __builtin_amdgcn_wmma_f32_16x16x32_bf16(
                         false, af, false, b0, (short)0, acc[0], false, false);
            acc[1] = __builtin_amdgcn_wmma_f32_16x16x32_bf16(
                         false, af, false, b1, (short)0, acc[1], false, false);
            __syncthreads();
        }
    } else {
        // ---- B staged transposed [n][k] bf16: lane reads 32B contiguous ----
        __shared__ alignas(32) __bf16 Bs2[TN][TKP];

        for (int k0 = 0; k0 < K; k0 += TK) {
            stageA(k0);
            #pragma unroll
            for (int i = 0; i < 2; ++i) {
                int idx = tid + i * BLK;        // 0..511
                int cc = idx & 63;              // n (coalesced)
                int kk = (idx >> 6) * 4;        // k group: 0,4,..,28
                const float* wp = &Wb[(long)(k0 + kk) * ldb + nBase + cc];
                v4bf wb4;
                wb4[0] = (__bf16)wp[0];
                wb4[1] = (__bf16)wp[(long)ldb];
                wb4[2] = (__bf16)wp[2l * ldb];
                wb4[3] = (__bf16)wp[3l * ldb];
                *(v4bf*)&Bs2[cc][kk] = wb4;
            }
            if (k0 + TK < K)
                __builtin_prefetch(&Wb[(long)(k0 + TK + (tid >> 6)) * ldb + nBase + (tid & 63)], 0, 3);
            __syncthreads();

            const v16bf af = *(const v16bf*)&As2[wm * 16 + ln][16 * half];
            const v16bf b0 = *(const v16bf*)&Bs2[wn * 32 + ln][16 * half];
            const v16bf b1 = *(const v16bf*)&Bs2[wn * 32 + 16 + ln][16 * half];

            acc[0] = __builtin_amdgcn_wmma_f32_16x16x32_bf16(
                         false, af, false, b0, (short)0, acc[0], false, false);
            acc[1] = __builtin_amdgcn_wmma_f32_16x16x32_bf16(
                         false, af, false, b1, (short)0, acc[1], false, false);
            __syncthreads();
        }
    }

    // ---- epilogue: C layout lane&15 = N, rows = 8*half + v ----
    const int mG0 = mBase + wm * 16 + half * 8;
    #pragma unroll
    for (int t = 0; t < 2; ++t) {
        const int nG = nBase + wn * 32 + t * 16 + ln;
        const float bv = bias ? bias[nG] : 0.f;
        #pragma unroll
        for (int v = 0; v < 8; ++v) {
            const int mG = mG0 + v;
            float val = acc[t][v];
            if constexpr (EPI == 2) {
                val = (nG <= mG) ? val * scale : -1e9f;
            } else {
                val += bv;
                if constexpr (EPI == 1)
                    val = 0.5f * val * (1.f + erff(val * 0.70710678118654752f));
                if constexpr (HAS_RES)
                    val += Rb[(long)mG * ldc + nG];
            }
            Cb[(long)mG * ldc + nG] = val;
        }
    }
}

__global__ __launch_bounds__(256) void layernorm_k(
    const float* __restrict__ x, const float* __restrict__ g,
    const float* __restrict__ b, float* __restrict__ y, int D)
{
    __shared__ float red[256];
    const long row = blockIdx.x;
    const float* xr = x + row * D;
    float s = 0.f;
    for (int i = threadIdx.x; i < D; i += 256) s += xr[i];
    red[threadIdx.x] = s; __syncthreads();
    for (int o = 128; o > 0; o >>= 1) {
        if (threadIdx.x < o) red[threadIdx.x] += red[threadIdx.x + o];
        __syncthreads();
    }
    const float mean = red[0] / D;
    __syncthreads();
    float v = 0.f;
    for (int i = threadIdx.x; i < D; i += 256) { float d = xr[i] - mean; v += d * d; }
    red[threadIdx.x] = v; __syncthreads();
    for (int o = 128; o > 0; o >>= 1) {
        if (threadIdx.x < o) red[threadIdx.x] += red[threadIdx.x + o];
        __syncthreads();
    }
    const float rstd = rsqrtf(red[0] / D + 1e-5f);
    float* yr = y + row * D;
    for (int i = threadIdx.x; i < D; i += 256)
        yr[i] = (xr[i] - mean) * rstd * g[i] + b[i];
}

__global__ __launch_bounds__(256) void softmax_k(float* __restrict__ p, int N)
{
    __shared__ float red[256];
    float* row = p + (long)blockIdx.x * N;
    float mx = -1e30f;
    for (int i = threadIdx.x; i < N; i += 256) mx = fmaxf(mx, row[i]);
    red[threadIdx.x] = mx; __syncthreads();
    for (int o = 128; o > 0; o >>= 1) {
        if (threadIdx.x < o) red[threadIdx.x] = fmaxf(red[threadIdx.x], red[threadIdx.x + o]);
        __syncthreads();
    }
    const float m = red[0];
    __syncthreads();
    float s = 0.f;
    for (int i = threadIdx.x; i < N; i += 256) {
        float e = __expf(row[i] - m); row[i] = e; s += e;
    }
    red[threadIdx.x] = s; __syncthreads();
    for (int o = 128; o > 0; o >>= 1) {
        if (threadIdx.x < o) red[threadIdx.x] += red[threadIdx.x + o];
        __syncthreads();
    }
    const float inv = 1.f / red[0];
    for (int i = threadIdx.x; i < N; i += 256) row[i] *= inv;
}

__global__ void embed_k(const int* __restrict__ tok, const float* __restrict__ emb,
                        const float* __restrict__ pe, float* __restrict__ x,
                        int T, int D, long total)
{
    long i = (long)blockIdx.x * blockDim.x + threadIdx.x;
    if (i >= total) return;
    int  d  = (int)(i % D);
    long bt = i / D;
    int  t  = (int)(bt % T);
    x[i] = emb[(long)tok[bt] * D + d] + pe[(long)t * D + d];
}

// ---------------------------------------------------------------------------
extern "C" void kernel_launch(void* const* d_in, const int* in_sizes, int n_in,
                              void* d_out, int out_size, void* d_ws, size_t ws_size,
                              hipStream_t stream)
{
    (void)in_sizes; (void)n_in; (void)out_size; (void)ws_size;
    constexpr int Bc = 2, Tc = 1024, Dc = 1024, Hc = 16, HDc = 64, Lc = 4,
                  Fc = 4096, Vc = 32000;

    const int*   tokens = (const int*)  d_in[0];
    /* d_in[1] = mask (causal; computed analytically) */
    const float* emb    = (const float*)d_in[2];
    const float* pe     = (const float*)d_in[3];
    const float* ln1_g  = (const float*)d_in[4];
    const float* ln1_b  = (const float*)d_in[5];
    const float* Wqkv   = (const float*)d_in[6];
    const float* bqkv   = (const float*)d_in[7];
    const float* Wproj  = (const float*)d_in[8];
    const float* bproj  = (const float*)d_in[9];
    const float* ln2_g  = (const float*)d_in[10];
    const float* ln2_b  = (const float*)d_in[11];
    const float* W1     = (const float*)d_in[12];
    const float* b1     = (const float*)d_in[13];
    const float* W2     = (const float*)d_in[14];
    const float* b2     = (const float*)d_in[15];
    const float* lnf_g  = (const float*)d_in[16];
    const float* lnf_b  = (const float*)d_in[17];
    const float* out_b  = (const float*)d_in[18];

    const long nTok = (long)Bc * Tc;              // 2048 rows

    float* ws     = (float*)d_ws;
    float* x      = ws;                           //  [2048,1024]
    float* lnbuf  = x     + nTok * Dc;            //  [2048,1024]
    float* qkv    = lnbuf + nTok * Dc;            //  [2048,3072]
    float* aout   = qkv   + nTok * 3 * Dc;        //  [2048,1024]
    float* h1     = aout  + nTok * Dc;            //  [2048,4096]

    float* logits = (float*)d_out;                       // [2048,32000]
    float* attnp  = logits + nTok * (long)Vc;            // [B,H,T,T] (last layer kept)

    // x = emb[tokens] + pe
    {
        long total = nTok * Dc;
        embed_k<<<(unsigned)((total + 255) / 256), 256, 0, stream>>>(
            tokens, emb, pe, x, Tc, Dc, total);
    }

    for (int l = 0; l < Lc; ++l) {
        // ln1
        layernorm_k<<<(unsigned)nTok, 256, 0, stream>>>(
            x, ln1_g + (long)l * Dc, ln1_b + (long)l * Dc, lnbuf, Dc);

        // qkv = ln1(x) @ Wqkv + bqkv   [2048,3072]
        gemm_wmma_k<0, false, false>
            <<<dim3(3 * Dc / TN, (unsigned)(nTok / TM), 1), BLK, 0, stream>>>(
            lnbuf, Wqkv + (long)l * Dc * 3 * Dc, bqkv + (long)l * 3 * Dc, nullptr, qkv,
            (int)nTok, 3 * Dc, Dc, Dc, 3 * Dc, 3 * Dc, 1,
            0, 0, 0, 0, 0, 0, 1.f);

        // scores[b,h] = mask(q @ k^T / sqrt(HD))  -> attnp (d_out region), TDM B-path
        gemm_wmma_k<2, true, false>
            <<<dim3(Tc / TN, Tc / TM, Bc * Hc), BLK, 0, stream>>>(
            qkv, qkv + Dc, nullptr, nullptr, attnp,
            Tc, Tc, HDc, 3 * Dc, 3 * Dc, Tc, Hc,
            (long)Tc * 3 * Dc, (long)HDc,          // A(q): per-b, per-h strides
            (long)Tc * 3 * Dc, (long)HDc,          // W(k)
            (long)Hc * Tc * Tc, (long)Tc * Tc,     // C(scores)
            0.125f);                               // 1/sqrt(64)

        // softmax rows in place
        softmax_k<<<(unsigned)(Bc * Hc * Tc), 256, 0, stream>>>(attnp, Tc);

        // aout[b,:,h*64:+64] = attn[b,h] @ v[b,h]
        gemm_wmma_k<0, false, false>
            <<<dim3(HDc / TN, Tc / TM, Bc * Hc), BLK, 0, stream>>>(
            attnp, qkv + 2 * Dc, nullptr, nullptr, aout,
            Tc, HDc, Tc, Tc, 3 * Dc, Dc, Hc,
            (long)Hc * Tc * Tc, (long)Tc * Tc,     // A(attn)
            (long)Tc * 3 * Dc, (long)HDc,          // W(v)
            (long)Tc * Dc, (long)HDc,              // C(aout)
            1.f);

        // x = x + aout @ Wproj + bproj
        gemm_wmma_k<0, false, true>
            <<<dim3(Dc / TN, (unsigned)(nTok / TM), 1), BLK, 0, stream>>>(
            aout, Wproj + (long)l * Dc * Dc, bproj + (long)l * Dc, x, x,
            (int)nTok, Dc, Dc, Dc, Dc, Dc, 1,
            0, 0, 0, 0, 0, 0, 1.f);

        // ln2
        layernorm_k<<<(unsigned)nTok, 256, 0, stream>>>(
            x, ln2_g + (long)l * Dc, ln2_b + (long)l * Dc, lnbuf, Dc);

        // h1 = gelu(ln2(x) @ W1 + b1)   [2048,4096]
        gemm_wmma_k<1, false, false>
            <<<dim3(Fc / TN, (unsigned)(nTok / TM), 1), BLK, 0, stream>>>(
            lnbuf, W1 + (long)l * Dc * Fc, b1 + (long)l * Fc, nullptr, h1,
            (int)nTok, Fc, Dc, Dc, Fc, Fc, 1,
            0, 0, 0, 0, 0, 0, 1.f);

        // x = x + h1 @ W2 + b2
        gemm_wmma_k<0, false, true>
            <<<dim3(Dc / TN, (unsigned)(nTok / TM), 1), BLK, 0, stream>>>(
            h1, W2 + (long)l * Fc * Dc, b2 + (long)l * Dc, x, x,
            (int)nTok, Dc, Fc, Fc, Dc, Dc, 1,
            0, 0, 0, 0, 0, 0, 1.f);
    }

    // final LN
    layernorm_k<<<(unsigned)nTok, 256, 0, stream>>>(x, lnf_g, lnf_b, lnbuf, Dc);

    // logits = lnf(x) @ emb^T + out_b   (weight tying; emb [V,D] -> TRANS_B, TDM B-path)
    gemm_wmma_k<0, true, false>
        <<<dim3(Vc / TN, (unsigned)(nTok / TM), 1), BLK, 0, stream>>>(
        lnbuf, emb, out_b, nullptr, logits,
        (int)nTok, Vc, Dc, Dc, Dc, Vc, 1,
        0, 0, 0, 0, 0, 0, 1.f);
}
